// GlmImageTextAttention_52725018526062
// MI455X (gfx1250) — matrix operand order repaired and checked by reading, not compiled, not run on hardware
//
#include <hip/hip_runtime.h>
#include <stdint.h>

typedef __attribute__((ext_vector_type(16))) __bf16 v16bf;
typedef __attribute__((ext_vector_type(8)))  float  v8f;
typedef unsigned int u32x4 __attribute__((ext_vector_type(4)));
typedef int          i32x8 __attribute__((ext_vector_type(8)));
typedef int          i32x4 __attribute__((ext_vector_type(4)));

#define HIDDEN  4096
#define NH      32
#define NKV     8
#define HD      128
#define QKV_OUT ((NH + 2 * NKV) * HD)   // 6144

union Frag {
    v16bf    v;
    uint32_t u[8];
    uint4    q[2];
};

union H8 {
    __bf16 h[8];
    uint4  q;
};

static __device__ __forceinline__ v8f vzero8() {
    v8f z = {0.f, 0.f, 0.f, 0.f, 0.f, 0.f, 0.f, 0.f};
    return z;
}

// ---------------------------------------------------------------------------
// TDM 2-D tile load: LDS[lds_addr] <= tile (tile_d0 x tile_d1) of 2-byte
// elements, row stride stride0 elements, with LDS padding (pad_amount_code+1
// dwords inserted every 2^(pad_interval_code+1) dwords) for bank-conflict-free
// fragment reads.  D# bitfields per CDNA5 ISA 8.3/8.4.
// ---------------------------------------------------------------------------
static __device__ __forceinline__ void tdm_load_2d(
    uint32_t lds_addr, const void* gptr,
    uint32_t tile_d0, uint32_t tile_d1, uint64_t stride0,
    uint32_t pad_interval_code, uint32_t pad_amount_code)
{
    uint64_t ga = (uint64_t)(uintptr_t)gptr;
    u32x4 g0;
    g0[0] = 1u;                                   // count=1, user descriptor
    g0[1] = lds_addr;                             // LDS byte address
    g0[2] = (uint32_t)ga;                         // global_addr[31:0]
    g0[3] = (uint32_t)(ga >> 32) | (2u << 30);    // global_addr[56:32] | type=2

    uint32_t w0 = (1u << 16)                      // data_size = 1 -> 2 bytes
                | (1u << 20)                      // pad_enable
                | (pad_interval_code << 22)
                | (pad_amount_code << 25);
    uint32_t td0 = tile_d0, td1 = tile_d1;        // tensor dims = tile dims
    i32x8 g1;
    g1[0] = (int)w0;
    g1[1] = (int)(td0 << 16);                             // [63:48] tensor_dim0 lo
    g1[2] = (int)((td0 >> 16) | (td1 << 16));             // dim0 hi | dim1 lo
    g1[3] = (int)((td1 >> 16) | (tile_d0 << 16));         // dim1 hi | tile_dim0
    g1[4] = (int)(tile_d1 & 0xFFFFu);                     // tile_dim1, tile_dim2=0
    g1[5] = (int)(uint32_t)stride0;                       // dim0_stride[31:0]
    g1[6] = (int)((uint32_t)(stride0 >> 32) & 0xFFFFu);   // dim0_stride[47:32]
    g1[7] = 0;

    i32x4 z4 = {0, 0, 0, 0};
    i32x8 z8 = {0, 0, 0, 0, 0, 0, 0, 0};
    __builtin_amdgcn_tensor_load_to_lds(g0, g1, z4, z4, z8, 0);
}

// ---------------------------------------------------------------------------
// Tiled bf16 WMMA GEMM with on-the-fly f32 -> bf16 conversion into LDS.
// C[M,N] = A[M,K] * B[K,N] (+ bias[N]).  M%128==0, N%128==0, K%32==0.
// 256 threads = 8 waves; wave tile 64x32 = 4x2 tiles of 16x16.
// ---------------------------------------------------------------------------
__global__ __launch_bounds__(256)
void gemm_bf16_wmma(const float* __restrict__ A, const float* __restrict__ B,
                    const float* __restrict__ bias, float* __restrict__ C,
                    int M, int N, int K)
{
    constexpr int LDS_STRIDE = 40;                 // halves; 80B rows keep b128 aligned
    __shared__ __bf16 As[128 * LDS_STRIDE];        // [m][k], k contiguous
    __shared__ __bf16 Bs[128 * LDS_STRIDE];        // [n][k], k contiguous (transposed)

    const int tid  = threadIdx.x;
    const int wid  = tid >> 5;
    const int lane = tid & 31;
    const int lo   = lane & 15;
    const int hi   = lane >> 4;
    const int wm   = wid >> 2;                     // 0..1
    const int wn   = wid & 3;                      // 0..3
    const int m0   = blockIdx.y * 128;
    const int n0   = blockIdx.x * 128;

    v8f acc[4][2];
#pragma unroll
    for (int i = 0; i < 4; i++)
#pragma unroll
        for (int j = 0; j < 2; j++) acc[i][j] = vzero8();

    for (int k0 = 0; k0 < K; k0 += 32) {
        // --- stage A tile 128x32 (f32 -> bf16), M-major ---
#pragma unroll
        for (int u = tid; u < 512; u += 256) {
            int row = u >> 2, seg = u & 3;
            const float4* gp = (const float4*)(A + (size_t)(m0 + row) * K + k0 + seg * 8);
            float4 f0 = gp[0], f1 = gp[1];
            H8 pk;
            pk.h[0] = (__bf16)f0.x; pk.h[1] = (__bf16)f0.y;
            pk.h[2] = (__bf16)f0.z; pk.h[3] = (__bf16)f0.w;
            pk.h[4] = (__bf16)f1.x; pk.h[5] = (__bf16)f1.y;
            pk.h[6] = (__bf16)f1.z; pk.h[7] = (__bf16)f1.w;
            *(uint4*)(&As[row * LDS_STRIDE + seg * 8]) = pk.q;
        }
        // --- stage B tile 32x128 transposed to [n][k] (f32 -> bf16) ---
#pragma unroll
        for (int u = tid; u < 512; u += 256) {
            int kr = u >> 4, cs = u & 15;
            const float4* gp = (const float4*)(B + (size_t)(k0 + kr) * N + n0 + cs * 8);
            float4 f0 = gp[0], f1 = gp[1];
            __bf16 h[8] = {(__bf16)f0.x, (__bf16)f0.y, (__bf16)f0.z, (__bf16)f0.w,
                           (__bf16)f1.x, (__bf16)f1.y, (__bf16)f1.z, (__bf16)f1.w};
            int nb = cs * 8;
#pragma unroll
            for (int e = 0; e < 8; e++) Bs[(nb + e) * LDS_STRIDE + kr] = h[e];
        }
        // --- prefetch next K-step tiles into cache ---
        if (k0 + 32 < K) {
            __builtin_prefetch(A + (size_t)(m0 + (tid >> 2)) * K + k0 + 32 + (tid & 3) * 8, 0, 1);
            __builtin_prefetch(B + (size_t)(k0 + 32 + (tid >> 4)) * N + n0 + (tid & 15) * 8, 0, 1);
        }
        __syncthreads();

        // --- fragments: contiguous k = 16*hi + s, same order for A and B ---
        Frag a[4], b[2];
#pragma unroll
        for (int i = 0; i < 4; i++) {
            const __bf16* p = &As[(wm * 64 + i * 16 + lo) * LDS_STRIDE];
#pragma unroll
            for (int v = 0; v < 8; v++) {
                int kp = 2 * v + 16 * hi;
                a[i].u[v] = *(const uint32_t*)(p + kp);
            }
        }
#pragma unroll
        for (int j = 0; j < 2; j++) {
            const __bf16* p = &Bs[(wn * 32 + j * 16 + lo) * LDS_STRIDE];
#pragma unroll
            for (int v = 0; v < 8; v++) {
                int kp = 2 * v + 16 * hi;
                b[j].u[v] = *(const uint32_t*)(p + kp);
            }
        }
#pragma unroll
        for (int i = 0; i < 4; i++)
#pragma unroll
            for (int j = 0; j < 2; j++)
                acc[i][j] = __builtin_amdgcn_wmma_f32_16x16x32_bf16(
                    false, a[i].v, false, b[j].v, (short)0, acc[i][j], false, false);
        __syncthreads();
    }

    // --- store C (+bias). C/D layout: lane -> N, vgpr r -> M = r + 8*hi ---
#pragma unroll
    for (int i = 0; i < 4; i++)
#pragma unroll
        for (int j = 0; j < 2; j++) {
            int col = n0 + wn * 32 + j * 16 + lo;
            float bv = bias ? bias[col] : 0.f;
#pragma unroll
            for (int r = 0; r < 8; r++) {
                int row = m0 + wm * 64 + i * 16 + r + 8 * hi;
                C[(size_t)row * N + col] = acc[i][j][r] + bv;
            }
        }
}

// ---------------------------------------------------------------------------
// MRoPE cos/sin tables: cos/sin[t][64] with sections [8,12,12] over positions[3][n]
// ---------------------------------------------------------------------------
__global__ void mrope_cos_sin_kernel(const int* __restrict__ positions,
                                     float* __restrict__ cosb, float* __restrict__ sinb, int n)
{
    int idx = blockIdx.x * blockDim.x + threadIdx.x;
    if (idx >= n * 32) return;
    int t = idx >> 5, j = idx & 31;
    int sel = (j < 8) ? 0 : (j < 20 ? 1 : 2);
    float pos = (float)positions[sel * n + t];
    float invf = expf(-((float)(2 * j) / 64.f) * logf(10000.f));
    float f = pos * invf;
    float c = cosf(f), s = sinf(f);
    cosb[t * 64 + j] = c; cosb[t * 64 + j + 32] = c;
    sinb[t * 64 + j] = s; sinb[t * 64 + j + 32] = s;
}

// ---------------------------------------------------------------------------
// RoPE apply on q,k (first 64 dims per head) + pack to bf16.
// q -> [t][32*128], k -> [t][8*128], v -> transposed [8*128][t]
// ---------------------------------------------------------------------------
__global__ __launch_bounds__(256)
void rope_pack_kernel(const float* __restrict__ qkv,
                      const float* __restrict__ cosb, const float* __restrict__ sinb,
                      __bf16* __restrict__ qb, __bf16* __restrict__ kb,
                      __bf16* __restrict__ vt, int n)
{
    int t = blockIdx.x;
    const float* row = qkv + (size_t)t * QKV_OUT;
    for (int idx = threadIdx.x; idx < QKV_OUT; idx += 256) {
        float x = row[idx];
        int d = idx & 127;
        float outv = x;
        if (idx < (NH + NKV) * HD && d < 64) {    // q or k rotated region
            int base = idx - d;
            if (d < 32)
                outv = x * cosb[t * 64 + d] - row[base + d + 32] * sinb[t * 64 + d];
            else
                outv = x * cosb[t * 64 + d] + row[base + d - 32] * sinb[t * 64 + d];
        }
        __bf16 h = (__bf16)outv;
        if (idx < NH * HD) {
            qb[(size_t)t * (NH * HD) + idx] = h;
        } else if (idx < (NH + NKV) * HD) {
            kb[(size_t)t * (NKV * HD) + (idx - NH * HD)] = h;
        } else {
            int r = idx - (NH + NKV) * HD;        // hv*128 + d
            vt[(size_t)r * n + t] = h;
        }
    }
}

// ---------------------------------------------------------------------------
// Causal flash attention, GQA 32q/8kv, head_dim 128.
// grid = (n/64 query blocks, 32 heads); 128 threads = 4 waves x 16 queries.
// K/V tiles double-buffered in LDS via the Tensor Data Mover (wave 0 issues,
// TENSORcnt synchronizes); TDM row padding makes ds_load_b128 fragment reads
// bank-conflict-free (K rows 272B = 68 words, V rows 80B = 20 words).
// ---------------------------------------------------------------------------
#define PLDS_SZ   (4 * 16 * 36 * 2)   // 4608, 256-aligned
#define KROW_B    272                 // 256B row + 16B TDM pad
#define VROW_B    80                  // 64B row + 16B TDM pad
#define KBUF_SZ   (32 * KROW_B)       // 8704
#define VBUF_SZ   (128 * VROW_B)      // 10240
#define KBUF_OFF  PLDS_SZ
#define VBUF_OFF  (KBUF_OFF + 2 * KBUF_SZ + 128)  // 22144: 128-aligned only
#define SMEM_SZ   (VBUF_OFF + 2 * VBUF_SZ)

__global__ __launch_bounds__(128)
void attn_kernel(const __bf16* __restrict__ qb, const __bf16* __restrict__ kb,
                 const __bf16* __restrict__ vt, float* __restrict__ ctx, int n)
{
    constexpr float SCALE = 0.08838834764831845f; // 1/sqrt(128)
    __shared__ __align__(256) char smem[SMEM_SZ];
    const uint32_t smem_base = (uint32_t)(uintptr_t)(&smem[0]); // flat lo32 = LDS addr

    const int h    = blockIdx.y;
    const int hkv  = h >> 2;                      // rep = 4
    const int qblk = blockIdx.x;
    const int wid  = threadIdx.x >> 5;
    const int lane = threadIdx.x & 31;
    const int lo   = lane & 15;
    const int hi   = lane >> 4;
    const int qbase = qblk * 64 + wid * 16;

    __bf16* pl = (__bf16*)(smem + wid * (16 * 36 * 2));

    // Q fragments: 16 rows x 128 K, 4 chunks of 32 (A-matrix layout)
    Frag aq[4];
    {
        const __bf16* qp = qb + (size_t)(qbase + lo) * (NH * HD) + h * HD;
#pragma unroll
        for (int i = 0; i < 4; i++) {
            int d0 = i * 32;
            aq[i].q[0] = *(const uint4*)(qp + d0 + 8 * hi);
            aq[i].q[1] = *(const uint4*)(qp + d0 + 16 + 8 * hi);
        }
    }

    float mrow[8], lrow[8];
#pragma unroll
    for (int r = 0; r < 8; r++) { mrow[r] = -1e30f; lrow[r] = 0.f; }
    v8f O[8];
#pragma unroll
    for (int j = 0; j < 8; j++) O[j] = vzero8();

    const int nblocks = qblk * 2 + 2;             // 32-key blocks up to causal edge

    // TDM prologue: block 0 into buffer 0 (K tile 32x128, V tile 128x32)
    if (wid == 0) {
        tdm_load_2d(smem_base + KBUF_OFF, kb + (size_t)0 * (NKV * HD) + hkv * HD,
                    128, 32, NKV * HD, /*interval 64dw*/5, /*pad 4dw*/3);
        tdm_load_2d(smem_base + VBUF_OFF, vt + (size_t)hkv * HD * n + 0,
                    32, 128, (uint64_t)n, /*interval 16dw*/3, /*pad 4dw*/3);
    }

    for (int kb2 = 0; kb2 < nblocks; kb2++) {
        const int kt0 = kb2 * 32;
        const int cur = kb2 & 1;

        if (wid == 0) {
            if (kb2 + 1 < nblocks) {
                const int nkt = (kb2 + 1) * 32;
                const int nb  = cur ^ 1;
                tdm_load_2d(smem_base + KBUF_OFF + nb * KBUF_SZ,
                            kb + (size_t)nkt * (NKV * HD) + hkv * HD,
                            128, 32, NKV * HD, 5, 3);
                tdm_load_2d(smem_base + VBUF_OFF + nb * VBUF_SZ,
                            vt + (size_t)hkv * HD * n + nkt,
                            32, 128, (uint64_t)n, 3, 3);
                __builtin_amdgcn_s_wait_tensorcnt(2);   // current block's 2 ops done
            } else {
                __builtin_amdgcn_s_wait_tensorcnt(0);
            }
        }
        __syncthreads();                                // release consumers

        const char* kl = smem + KBUF_OFF + cur * KBUF_SZ;
        const char* vl = smem + VBUF_OFF + cur * VBUF_SZ;

        // ---- S = Q * K^T (16x32) ----
        v8f S[2] = {vzero8(), vzero8()};
#pragma unroll
        for (int i = 0; i < 4; i++) {
            int d0 = i * 32;
#pragma unroll
            for (int j = 0; j < 2; j++) {
                Frag bk;
                const __bf16* kp = (const __bf16*)(kl + (j * 16 + lo) * KROW_B
                                                      + (d0 + 8 * hi) * 2);
                bk.q[0] = *(const uint4*)(kp);
                bk.q[1] = *(const uint4*)(kp + 16);
                S[j] = __builtin_amdgcn_wmma_f32_16x16x32_bf16(
                    false, aq[i].v, false, bk.v, (short)0, S[j], false, false);
            }
        }

        // ---- scale + causal mask + online softmax ----
        float mnew[8];
#pragma unroll
        for (int r = 0; r < 8; r++) {
            int qt = qbase + r + 8 * hi;
            float s0 = (kt0 + lo      <= qt) ? S[0][r] * SCALE : -1e30f;
            float s1 = (kt0 + 16 + lo <= qt) ? S[1][r] * SCALE : -1e30f;
            S[0][r] = s0; S[1][r] = s1;
            float mx = fmaxf(s0, s1);
#pragma unroll
            for (int d = 8; d >= 1; d >>= 1) mx = fmaxf(mx, __shfl_xor(mx, d, 32));
            mnew[r] = fmaxf(mrow[r], mx);
        }
#pragma unroll
        for (int r = 0; r < 8; r++) {
            float p0 = __expf(S[0][r] - mnew[r]);
            float p1 = __expf(S[1][r] - mnew[r]);
            S[0][r] = p0; S[1][r] = p1;
            float sum = p0 + p1;
#pragma unroll
            for (int d = 8; d >= 1; d >>= 1) sum += __shfl_xor(sum, d, 32);
            float alpha = __expf(mrow[r] - mnew[r]);
            lrow[r] = lrow[r] * alpha + sum;
            mrow[r] = mnew[r];
#pragma unroll
            for (int j = 0; j < 8; j++) O[j][r] *= alpha;
        }

        // ---- re-layout P (C layout f32) -> A layout bf16 via per-wave LDS ----
#pragma unroll
        for (int r = 0; r < 8; r++) {
            int row = r + 8 * hi;
            pl[row * 36 + lo]      = (__bf16)S[0][r];
            pl[row * 36 + 16 + lo] = (__bf16)S[1][r];
        }
        asm volatile("s_wait_dscnt 0" ::: "memory");
        Frag ap;
        {
            const __bf16* pp = pl + lo * 36;
#pragma unroll
            for (int v = 0; v < 8; v++) {
                int kp = 2 * v + 16 * hi;
                ap.u[v] = *(const uint32_t*)(pp + kp);
            }
        }

        // ---- O += P * V  (V tile in LDS, keys contiguous along rows) ----
#pragma unroll
        for (int j = 0; j < 8; j++) {
            Frag bv;
            const __bf16* vp = (const __bf16*)(vl + (j * 16 + lo) * VROW_B + 32 * hi);
            bv.q[0] = *(const uint4*)(vp);
            bv.q[1] = *(const uint4*)(vp + 8);
            O[j] = __builtin_amdgcn_wmma_f32_16x16x32_bf16(
                false, ap.v, false, bv.v, (short)0, O[j], false, false);
        }
        __syncthreads();                                // before buffer reuse
    }

    // ---- epilogue: O / l -> ctx f32 [t][32*128] ----
#pragma unroll
    for (int r = 0; r < 8; r++) {
        float inv = 1.f / lrow[r];
        int row = qbase + r + 8 * hi;
        float* cp = ctx + (size_t)row * (NH * HD) + h * HD;
#pragma unroll
        for (int j = 0; j < 8; j++) cp[j * 16 + lo] = O[j][r] * inv;
    }
}

// ---------------------------------------------------------------------------
extern "C" void kernel_launch(void* const* d_in, const int* in_sizes, int n_in,
                              void* d_out, int out_size, void* d_ws, size_t ws_size,
                              hipStream_t stream)
{
    const float* hidden    = (const float*)d_in[0];
    const float* w_qkv     = (const float*)d_in[1];
    const float* b_qkv     = (const float*)d_in[2];
    const float* w_o       = (const float*)d_in[3];
    const int*   positions = (const int*)d_in[4];
    const int n = in_sizes[0] / HIDDEN;            // 2048

    char* ws = (char*)d_ws;
    size_t off = 0;
    auto alloc = [&](size_t bytes) -> void* {
        void* p = ws + off;
        off += (bytes + 255) & ~(size_t)255;
        return p;
    };
    float*  qkv_f32 = (float*) alloc((size_t)n * QKV_OUT * 4);
    float*  cosb    = (float*) alloc((size_t)n * 64 * 4);
    float*  sinb    = (float*) alloc((size_t)n * 64 * 4);
    __bf16* q_bf    = (__bf16*)alloc((size_t)n * NH * HD * 2);
    __bf16* k_bf    = (__bf16*)alloc((size_t)n * NKV * HD * 2);
    __bf16* v_t     = (__bf16*)alloc((size_t)NKV * HD * n * 2);
    float*  ctx     = (float*) alloc((size_t)n * NH * HD * 4);

    // 1) QKV projection (+bias)
    gemm_bf16_wmma<<<dim3(QKV_OUT / 128, n / 128), 256, 0, stream>>>(
        hidden, w_qkv, b_qkv, qkv_f32, n, QKV_OUT, HIDDEN);

    // 2) MRoPE tables
    mrope_cos_sin_kernel<<<(n * 32 + 255) / 256, 256, 0, stream>>>(positions, cosb, sinb, n);

    // 3) RoPE apply + bf16 pack (q, k row-major; v transposed)
    rope_pack_kernel<<<n, 256, 0, stream>>>(qkv_f32, cosb, sinb, q_bf, k_bf, v_t, n);

    // 4) causal flash attention (TDM-staged K/V)
    attn_kernel<<<dim3(n / 64, NH), 128, 0, stream>>>(q_bf, k_bf, v_t, ctx, n);

    // 5) output projection
    gemm_bf16_wmma<<<dim3(HIDDEN / 128, n / 128), 256, 0, stream>>>(
        ctx, w_o, nullptr, (float*)d_out, n, HIDDEN, HIDDEN);
}
